// KBEmbedderAll_22497038696566
// MI455X (gfx1250) — compile-verified, matrix operand order checked by reading, not run on hardware
//
#include <hip/hip_runtime.h>
#include <hip/hip_bf16.h>

// Problem constants (from reference)
#define NN    16384
#define CC    16
#define DIM   300
#define DSPAN 512
#define HH    150
#define VV    200000

#define KA   512      // span-K
#define KB   320      // candidate-K padded 300 -> 320
#define KTOT 832      // KA + KB  (26 k-tiles of 32)
#define HP   160      // H padded 150 -> 160 (10 n-tiles of 16)
#define NKT  26       // KTOT/32
#define NNT  10       // HP/16

typedef __bf16 v16bf __attribute__((ext_vector_type(16)));
typedef float  v8f   __attribute__((ext_vector_type(8)));
typedef unsigned int v4u __attribute__((ext_vector_type(4)));
typedef int v8i __attribute__((ext_vector_type(8)));
typedef int v4i __attribute__((ext_vector_type(4)));

// K index held in element i of a 16-bit A/B fragment (CDNA5 ISA §7.12.2):
// lanes 0-15 hold K {0..7, 16..23}; lanes 16-31 hold K {8..15, 24..31}.
__device__ __forceinline__ int kmap(int i, int lane) {
  return (i < 8 ? i : i + 8) + ((lane & 16) ? 8 : 0);
}

// ---------------------------------------------------------------------------
// k0: repack W1 into WMMA-fragment order:
//   Bpk[((ktile*10 + ntile)*32 + lane)*16 + i] = bf16(W1[ktile*32+kmap(i,lane),
//                                                       ntile*16+(lane&15)])
// so a B fragment is one contiguous, 32B-aligned v16bf load per lane.
// Zero-padded beyond [812,150]. Also pad b1, W2 to fp32[160].
// ---------------------------------------------------------------------------
#define NBPK (NKT * NNT * 32 * 16)   // 133,120 bf16 elements

__global__ void k0_pack(const float* __restrict__ W1, const float* __restrict__ b1,
                        const float* __restrict__ W2, __bf16* __restrict__ Bpk,
                        float* __restrict__ b1p, float* __restrict__ W2p) {
  int gid = blockIdx.x * blockDim.x + threadIdx.x;
  if (gid < NBPK) {
    int i     = gid & 15;
    int lane  = (gid >> 4) & 31;
    int frag  = gid >> 9;
    int ntile = frag % NNT;
    int ktile = frag / NNT;
    int row = ktile * 32 + kmap(i, lane);
    int col = ntile * 16 + (lane & 15);
    float v = (row < (DSPAN + DIM) && col < HH) ? W1[row * HH + col] : 0.0f;
    Bpk[gid] = (__bf16)v;
  } else if (gid < NBPK + HP) {
    int j = gid - NBPK;
    b1p[j] = (j < HH) ? b1[j] : 0.0f;
  } else if (gid < NBPK + 2 * HP) {
    int j = gid - NBPK - HP;
    W2p[j] = (j < HH) ? W2[j] : 0.0f;   // W2 is [150,1]
  }
}

// ---------------------------------------------------------------------------
// k1: hspanT[col][n] = (sv[n,0:512] @ W1[0:512,:] + b1)[col]   (pre-activation)
// 16 spans per block, 10 waves (one per 16-col H tile), 16 K-steps of
// v_wmma_f32_16x16x32_bf16, C seeded with b1. Transposed store.
// ---------------------------------------------------------------------------
__global__ void k1_hspan(const float* __restrict__ sv, const __bf16* __restrict__ Bpk,
                         const float* __restrict__ b1p, float* __restrict__ hspanT) {
  __shared__ __bf16 Asv[16][KA];          // 16 KB
  const int tid = threadIdx.x;
  const int s0  = blockIdx.x * 16;

  for (int idx = tid; idx < 16 * KA; idx += 320) {
    int m = idx >> 9, k = idx & 511;
    Asv[m][k] = (__bf16)sv[(s0 + m) * DSPAN + k];
  }
  __syncthreads();

  const int wid  = tid >> 5;
  const int lane = tid & 31;
  const int col  = wid * 16 + (lane & 15);   // H column
  const int m    = lane & 15;                // A row (span within block)
  const int mhi  = (lane & 16) ? 8 : 0;      // C/D row offset
  const v16bf* Bp = (const v16bf*)Bpk;

  float b1v = b1p[col];
  v8f acc = {b1v, b1v, b1v, b1v, b1v, b1v, b1v, b1v};

#pragma unroll 4
  for (int k = 0; k < KA / 32; ++k) {
    v16bf a;
#pragma unroll
    for (int i = 0; i < 16; ++i)
      a[i] = Asv[m][k * 32 + kmap(i, lane)];
    v16bf b = Bp[(k * NNT + wid) * 32 + lane];
    acc = __builtin_amdgcn_wmma_f32_16x16x32_bf16(false, a, false, b,
                                                  (short)0, acc, false, false);
  }
  float* p = hspanT + col * NN + s0 + mhi;   // contiguous per lane -> b128 x2
#pragma unroll
  for (int r = 0; r < 8; ++r) p[r] = acc[r];
}

// ---------------------------------------------------------------------------
// k2: per span: TDM-gather 16 candidate embedding rows to LDS (tensor gather
// mode, 32-bit indices, 8 rows per descriptor -> 2 DMAs), bf16 convert,
// h = relu(hspan + cand@W1_bot) via WMMA, scores, masked softmax, weighted
// sum, output. 8 spans per block; wave w owns H-tile w and keeps its 10
// B-fragments in registers across all 8 spans.
// ---------------------------------------------------------------------------
__global__ void __launch_bounds__(320)
k2_main(const float* __restrict__ sv, const int* __restrict__ cand,
        const float* __restrict__ maskc, const float* __restrict__ embed,
        const __bf16* __restrict__ Bpk, const float* __restrict__ W2p,
        const float* __restrict__ b2, const float* __restrict__ hspanT,
        float* __restrict__ out) {
  __shared__ float  cand_f [16][DIM];      // 19,200 B (TDM gather target)
  __shared__ __bf16 cand_bf[16][KB];       // 10,240 B
  __shared__ float  h_sh   [16][168];      // 10,752 B
  __shared__ float  red_sh [16][17];       //  1,088 B
  __shared__ float  sc_sh[16], ex_sh[16], p_sh[16];

  const int tid  = threadIdx.x;
  const int lane = tid & 31;
  const int wid  = tid >> 5;
  const int col  = wid * 16 + (lane & 15);
  const int m    = lane & 15;
  const int mhi  = (lane & 16) ? 8 : 0;
  const v16bf* Bp = (const v16bf*)Bpk;

  // Preload candidate-half B fragments (ktiles 16..25) into registers.
  v16bf Bfrag[KB / 32];
#pragma unroll
  for (int k = 0; k < KB / 32; ++k)
    Bfrag[k] = Bp[((16 + k) * NNT + wid) * 32 + lane];

  const float b2v = b2[0];
  const int n0 = blockIdx.x * 8;

  // Invariant parts of the Tensor-DMA descriptor (CDNA5 ISA)
  // group0: count=1 | gather_index_size=32b (bit30) | gather_mode (bit31);
  //         lds_addr; global_addr(57b); type=2 in bits 127:126.
  // group1: data_size=4B (code 2, bits 17:16); tensor_dim0=300 (bits 79:48);
  //         tensor_dim1=V (bits 111:80); tile_dim0=300 (bits 127:112);
  //         tile_dim1=#indices=8 (bits 143:128); dim0_stride=300 (bits 207:160).
  const unsigned lds0 = (unsigned)(size_t)(&cand_f[0][0]);
  const unsigned long long ga = (unsigned long long)(size_t)embed;
  const v8i g1 = { (int)0x00020000u,
                   (int)(300u << 16),                    // tensor_dim0 lo16
                   (int)(((unsigned)VV & 0xFFFFu) << 16),// tensor_dim1 lo16
                   (int)((300u << 16) | ((unsigned)VV >> 16)), // tile_dim0 | dim1 hi16
                   8,                                    // tile_dim1 = 8 indices
                   300, 0, 0 };                          // tensor_dim0_stride = 300
  const v8i gz = { 0, 0, 0, 0, 0, 0, 0, 0 };             // unused trailing group

  for (int s = 0; s < 8; ++s) {
    const int n = n0 + s;

    // --- TDM gather: 16 embedding rows (f32) -> cand_f, wave 0 issues ---
    if (wid == 0) {
      int r[16];
#pragma unroll
      for (int c = 0; c < 16; ++c)
        r[c] = __builtin_amdgcn_readfirstlane(cand[n * CC + c]);
      v4u g0a = { 0xC0000001u, lds0, (unsigned)ga,
                  (unsigned)(ga >> 32) | 0x80000000u };
      v4u g0b = g0a;
      g0b[1] = lds0 + 8u * DIM * 4u;            // rows 8..15 land after rows 0..7
      v4i g2a = { r[0], r[1], r[2],  r[3]  };
      v4i g3a = { r[4], r[5], r[6],  r[7]  };
      v4i g2b = { r[8], r[9], r[10], r[11] };
      v4i g3b = { r[12], r[13], r[14], r[15] };
      __builtin_amdgcn_tensor_load_to_lds(g0a, g1, g2a, g3a, gz, 0);
      __builtin_amdgcn_tensor_load_to_lds(g0b, g1, g2b, g3b, gz, 0);
      __builtin_amdgcn_s_wait_tensorcnt(0);
    }
    __syncthreads();

    // bf16 copy of gathered rows (vectorized LDS->LDS), K-pad 300..319 = 0.
    for (int idx = tid; idx < 16 * (DIM / 4); idx += 320) {  // 1200 float4s
      int c = idx / 75, q = idx - c * 75;
      float4 v = *(const float4*)&cand_f[c][q * 4];
      __bf16* dst = &cand_bf[c][q * 4];
      dst[0] = (__bf16)v.x; dst[1] = (__bf16)v.y;
      dst[2] = (__bf16)v.z; dst[3] = (__bf16)v.w;
    }
    cand_bf[tid / 20][DIM + (tid % 20)] = (__bf16)0.0f;
    __syncthreads();

    // C seeded with shared span contribution (includes b1).
    float hs = hspanT[col * NN + n];
    v8f acc = {hs, hs, hs, hs, hs, hs, hs, hs};
#pragma unroll
    for (int k = 0; k < KB / 32; ++k) {
      v16bf a;
#pragma unroll
      for (int i = 0; i < 16; ++i)
        a[i] = cand_bf[m][k * 32 + kmap(i, lane)];
      acc = __builtin_amdgcn_wmma_f32_16x16x32_bf16(false, a, false, Bfrag[k],
                                                    (short)0, acc, false, false);
    }
#pragma unroll
    for (int r = 0; r < 8; ++r) {
      float h = acc[r];
      h_sh[r + mhi][col] = h > 0.0f ? h : 0.0f;         // relu
    }
    __syncthreads();

    // scores = h @ W2 + b2: 256 threads do strided partials, 16-way reduce.
    if (tid < 256) {
      int mm = tid >> 4, j0 = tid & 15;
      float part = 0.0f;
      for (int j = j0; j < HP; j += 16) part += h_sh[mm][j] * W2p[j];
      red_sh[mm][j0] = part;
    }
    __syncthreads();
    if (tid < 16) {
      float sc = b2v;
#pragma unroll
      for (int j = 0; j < 16; ++j) sc += red_sh[tid][j];
      if (maskc[n * CC + tid] == 0.0f) sc = -1e30f;     // log(0) mask
      sc_sh[tid] = sc;
    }
    __syncthreads();
    if (tid < 16) {
      float mx = sc_sh[0];
#pragma unroll
      for (int j = 1; j < 16; ++j) mx = fmaxf(mx, sc_sh[j]);
      ex_sh[tid] = __expf(sc_sh[tid] - mx);
    }
    __syncthreads();
    if (tid < 16) {
      float sum = 0.0f;
#pragma unroll
      for (int j = 0; j < 16; ++j) sum += ex_sh[j];
      p_sh[tid] = ex_sh[tid] / sum;
    }
    __syncthreads();

    // weighted candidate sum into out[.., 512:812], span copy into out[.., 0:512]
    if (tid < DIM) {
      float w = 0.0f;
#pragma unroll
      for (int c = 0; c < 16; ++c) w += p_sh[c] * cand_f[c][tid];
      out[n * (DSPAN + DIM) + DSPAN + tid] = w;
    }
    for (int d = tid; d < DSPAN; d += 320)
      out[n * (DSPAN + DIM) + d] = sv[n * DSPAN + d];
    __syncthreads();   // cand_f reads done before next span's TDM overwrites
  }
}

// ---------------------------------------------------------------------------
extern "C" void kernel_launch(void* const* d_in, const int* in_sizes, int n_in,
                              void* d_out, int out_size, void* d_ws, size_t ws_size,
                              hipStream_t stream) {
  const float* sv    = (const float*)d_in[0];   // [1, N, 512]
  const int*   cand  = (const int*)  d_in[1];   // [1, N, 16]
  const float* maskc = (const float*)d_in[2];   // [1, N, 16]
  const float* embed = (const float*)d_in[3];   // [V, 300]
  const float* W1    = (const float*)d_in[4];   // [812, 150]
  const float* b1    = (const float*)d_in[5];   // [150]
  const float* W2    = (const float*)d_in[6];   // [150, 1]
  const float* b2    = (const float*)d_in[7];   // [1]
  float* out = (float*)d_out;                   // [1, N, 812] f32

  // Workspace layout (~10.75 MB total):
  //   Bpk    : bf16, fragment-packed W1, 26*10*32*16   266,240 B @ 0
  //   b1p    : f32  [160]                                  640 B @ 266240
  //   W2p    : f32  [160]                                  640 B @ 266880
  //   hspanT : f32  [160][16384] (transposed)       10,485,760 B @ 267520
  char* ws = (char*)d_ws;
  __bf16* Bpk    = (__bf16*)(ws);
  float*  b1p    = (float*)(ws + 266240);
  float*  W2p    = (float*)(ws + 266880);
  float*  hspanT = (float*)(ws + 267520);

  k0_pack <<<(NBPK + 2 * HP + 255) / 256, 256, 0, stream>>>(W1, b1, W2, Bpk, b1p, W2p);
  k1_hspan<<<NN / 16, 320, 0, stream>>>(sv, Bpk, b1p, hspanT);
  k2_main <<<NN / 8, 320, 0, stream>>>(sv, cand, maskc, embed, Bpk, W2p, b2, hspanT, out);
}
